// EMAQuantizer_76716705841361
// MI455X (gfx1250) — matrix operand order repaired and checked by reading, not compiled, not run on hardware
//
#include <hip/hip_runtime.h>
#include <hip/hip_bf16.h>

typedef __attribute__((ext_vector_type(16))) __bf16 bf16x16;
typedef __attribute__((ext_vector_type(8)))  float  f32x8;
typedef int v4i __attribute__((vector_size(16)));   // matches async builtin param

#define DECAY_F  0.99f
#define OMD_F    0.01f      // 1 - DECAY
#define EPS_F    1e-5f
#define D_DIM    64
#define ROW_PAD  72         // bf16 elems per staged code row (bank-conflict pad)

#define AS1 __attribute__((address_space(1)))
#define AS3 __attribute__((address_space(3)))

#if defined(__has_builtin)
#if __has_builtin(__builtin_amdgcn_global_load_async_to_lds_b128)
#define USE_ASYNC_LDS 1
#endif
#endif

__device__ __forceinline__ void async_copy16(const __bf16* g, __bf16* l) {
#ifdef USE_ASYNC_LDS
  __builtin_amdgcn_global_load_async_to_lds_b128((AS1 v4i*)g, (AS3 v4i*)l, 0, 0);
#else
  *(float4*)l = *(const float4*)g;       // synchronous fallback
#endif
}

__device__ __forceinline__ void async_wait() {
#ifdef USE_ASYNC_LDS
#if __has_builtin(__builtin_amdgcn_s_wait_asynccnt)
  __builtin_amdgcn_s_wait_asynccnt(0);
#else
  asm volatile("s_wait_asynccnt 0x0" ::: "memory");
#endif
#endif
}

// ---------------------------------------------------------------- init ------
__global__ void vq_init(float* __restrict__ p, int n) {
  int i = blockIdx.x * blockDim.x + threadIdx.x;
  if (i < n) p[i] = 0.0f;
}

// ---------------------------------------------------------------- prep ------
// One wave per code k: split (-2 * embed_w[k,:]) into bf16 hi/lo and compute
// exact fp32 ||e_k||^2.
__global__ void __launch_bounds__(256) vq_prep(const float* __restrict__ embed_w,
                                               __bf16* __restrict__ ewHi,
                                               __bf16* __restrict__ ewLo,
                                               float* __restrict__ codeNorm,
                                               int K) {
  const int lane = threadIdx.x & 31;
  const int k    = blockIdx.x * 8 + (threadIdx.x >> 5);
  if (k >= K) return;
  const size_t base = (size_t)k * D_DIM + lane * 2;
  const float2 v = *(const float2*)(embed_w + base);
  float nrm = v.x * v.x + v.y * v.y;
  const float sx = -2.0f * v.x, sy = -2.0f * v.y;
  const __bf16 hx = (__bf16)sx, hy = (__bf16)sy;
  ewHi[base]     = hx;
  ewHi[base + 1] = hy;
  ewLo[base]     = (__bf16)(sx - (float)hx);
  ewLo[base + 1] = (__bf16)(sy - (float)hy);
  #pragma unroll
  for (int m = 16; m >= 1; m >>= 1) nrm += __shfl_xor(nrm, m, 32);
  if (lane == 0) codeNorm[k] = nrm;
}

// ---------------------------------------------------------------- main ------
// 256 threads = 8 waves; each wave owns 16 rows of flat z_e. The 8 waves scan
// the codebook in lock-step 16-column chunks; each chunk's bf16 hi/lo tile is
// async-staged into double-buffered LDS and shared by all waves.
__global__ void __launch_bounds__(256) vq_main(const float* __restrict__ z_e,
                                               const float* __restrict__ embed_w,
                                               const __bf16* __restrict__ ewHi,
                                               const __bf16* __restrict__ ewLo,
                                               const float* __restrict__ codeNorm,
                                               float* __restrict__ out_zq,
                                               float* __restrict__ out_ind,
                                               float* __restrict__ onehot,
                                               float* __restrict__ embed_sum,
                                               float* __restrict__ sqsum,
                                               int K) {
  __shared__ __bf16 sHi[2][16 * ROW_PAD];
  __shared__ __bf16 sLo[2][16 * ROW_PAD];
  __shared__ float  sNorm[1024];

  const int tid  = threadIdx.x;
  const int lane = tid & 31;
  const int wave = tid >> 5;
  const int row0 = (blockIdx.x * 8 + wave) * 16;
  const int h    = lane >> 4;     // lane half
  const int mr   = lane & 15;     // M for A-frag / N for B-frag

  // stage code norms once
  for (int i = tid; i < K; i += 256) sNorm[i] = codeNorm[i];

  // B-chunk staging assignment: threads 0-127 copy hi, 128-255 copy lo;
  // each thread moves 16B (8 bf16) of the 16x64 chunk into the padded buffer.
  const int hsel = tid >> 7;
  const int t    = tid & 127;
  const int crow = t >> 3;          // 0..15 (code row within chunk)
  const int ccol = (t & 7) * 8;     // bf16 column, 16B granules
  const __bf16* __restrict__ srcBase = hsel ? ewLo : ewHi;
  const int dstOff = crow * ROW_PAD + ccol;

  // ---- load A fragments (16 rows x 64 K), bf16 hi + lo, two 32-deep chunks
  bf16x16 aHi[2], aLo[2];
  #pragma unroll
  for (int kc = 0; kc < 2; ++kc) {
    #pragma unroll
    for (int p = 0; p < 8; ++p) {
      const int kb = (p < 4) ? (2 * p + 8 * h) : (16 + 2 * (p - 4) + 8 * h);
      const int k  = kc * 32 + kb;
      const float2 v = *(const float2*)(z_e + (size_t)(row0 + mr) * D_DIM + k);
      const __bf16 hx = (__bf16)v.x, hy = (__bf16)v.y;
      aHi[kc][2 * p]     = hx;
      aHi[kc][2 * p + 1] = hy;
      aLo[kc][2 * p]     = (__bf16)(v.x - (float)hx);
      aLo[kc][2 * p + 1] = (__bf16)(v.y - (float)hy);
    }
  }

  float minv[8];
  int   mini[8];
  #pragma unroll
  for (int r = 0; r < 8; ++r) { minv[r] = 3.4e38f; mini[r] = 0; }

  const int C = K >> 4;
  // prologue: stage chunk 0 into buffer 0
  {
    __bf16* dst = (hsel ? sLo[0] : sHi[0]) + dstOff;
    async_copy16(srcBase + (size_t)crow * D_DIM + ccol, dst);
  }

  for (int c = 0; c < C; ++c) {
    const int buf = c & 1;
    async_wait();        // this thread's copy into buf done (ASYNCcnt in order)
    __syncthreads();     // all copies visible; prior reads of other buf done

    if (c + 1 < C) {     // prefetch next chunk into the other buffer
      __bf16* dst = (hsel ? sLo[buf ^ 1] : sHi[buf ^ 1]) + dstOff;
      async_copy16(srcBase + (size_t)((c + 1) * 16 + crow) * D_DIM + ccol, dst);
    }

    // ---- read B fragments from LDS (table k order, matches A frags)
    const __bf16* __restrict__ bh = &sHi[buf][mr * ROW_PAD];
    const __bf16* __restrict__ bl = &sLo[buf][mr * ROW_PAD];
    bf16x16 bHi[2], bLo[2];
    #pragma unroll
    for (int kc = 0; kc < 2; ++kc) {
      #pragma unroll
      for (int v = 0; v < 8; ++v) {
        const int k = kc * 32 + ((v < 4) ? (2 * v + 8 * h) : (16 + 2 * (v - 4) + 8 * h));
        bHi[kc][2 * v]     = bh[k];
        bHi[kc][2 * v + 1] = bh[k + 1];
        bLo[kc][2 * v]     = bl[k];
        bLo[kc][2 * v + 1] = bl[k + 1];
      }
    }

    f32x8 acc = {0.f, 0.f, 0.f, 0.f, 0.f, 0.f, 0.f, 0.f};
    acc = __builtin_amdgcn_wmma_f32_16x16x32_bf16(false, aHi[0], false, bHi[0], (short)0, acc, false, false);
    acc = __builtin_amdgcn_wmma_f32_16x16x32_bf16(false, aHi[1], false, bHi[1], (short)0, acc, false, false);
    acc = __builtin_amdgcn_wmma_f32_16x16x32_bf16(false, aLo[0], false, bHi[0], (short)0, acc, false, false);
    acc = __builtin_amdgcn_wmma_f32_16x16x32_bf16(false, aLo[1], false, bHi[1], (short)0, acc, false, false);
    acc = __builtin_amdgcn_wmma_f32_16x16x32_bf16(false, aHi[0], false, bLo[0], (short)0, acc, false, false);
    acc = __builtin_amdgcn_wmma_f32_16x16x32_bf16(false, aHi[1], false, bLo[1], (short)0, acc, false, false);

    const int   n  = c * 16 + mr;
    const float cn = sNorm[n];            // acc holds -2*f.e
    #pragma unroll
    for (int r = 0; r < 8; ++r) {
      const float s = cn + acc[r];        // dist - ||f||^2 (row-constant)
      if (s < minv[r]) { minv[r] = s; mini[r] = n; }
    }
  }

  // ---- reduce min/argmin across the 16 lanes holding the same row
  #pragma unroll
  for (int m = 8; m >= 1; m >>= 1) {
    #pragma unroll
    for (int r = 0; r < 8; ++r) {
      const float ov = __shfl_xor(minv[r], m, 32);
      const int   oi = __shfl_xor(mini[r], m, 32);
      if (ov < minv[r] || (ov == minv[r] && oi < mini[r])) {
        minv[r] = ov; mini[r] = oi;
      }
    }
  }
  // lanes 0 / 16 now hold argmin for rows r / r+8

  // ---- emit z_q_st, indices, loss partial, EMA scatter-adds
  float sq = 0.0f;
  #pragma unroll
  for (int m = 0; m < 16; ++m) {
    const int idx = __shfl(mini[m & 7], (m < 8) ? 0 : 16, 32);
    const int d0  = lane * 2;
    const float2 e = *(const float2*)(embed_w + (size_t)idx * D_DIM + d0);
    const float2 z = *(const float2*)(z_e + (size_t)(row0 + m) * D_DIM + d0);
    const float tx = e.x - z.x, ty = e.y - z.y;    // z_q - z_e
    float2 o; o.x = z.x + tx; o.y = z.y + ty;      // straight-through
    *(float2*)(out_zq + (size_t)(row0 + m) * D_DIM + d0) = o;
    sq += tx * tx + ty * ty;
    atomicAdd(embed_sum + (size_t)d0 * K + idx, z.x);        // segment_sum^T [D,K]
    atomicAdd(embed_sum + (size_t)(d0 + 1) * K + idx, z.y);
    if (lane == 0) {
      atomicAdd(onehot + idx, 1.0f);
      out_ind[row0 + m] = (float)idx;
    }
  }
  #pragma unroll
  for (int m = 16; m >= 1; m >>= 1) sq += __shfl_xor(sq, m, 32);
  if (lane == 0) atomicAdd(sqsum, sq);
}

// ------------------------------------------------------------- finalize -----
__global__ void __launch_bounds__(1024) vq_finalize(const float* __restrict__ cluster_size,
                                                    const float* __restrict__ embed_avg,
                                                    const float* __restrict__ onehot,
                                                    const float* __restrict__ embed_sum,
                                                    const float* __restrict__ sqsum,
                                                    float* __restrict__ out_cs,
                                                    float* __restrict__ out_ea,
                                                    float* __restrict__ out_ew,
                                                    float* __restrict__ out_diff,
                                                    int K, float inv_count) {
  __shared__ float red[1024];
  const int k = threadIdx.x;
  const float ncs = cluster_size[k] * DECAY_F + OMD_F * onehot[k];
  out_cs[k] = ncs;
  red[k] = ncs;
  __syncthreads();
  for (int s = blockDim.x >> 1; s > 0; s >>= 1) {
    if (k < s) red[k] += red[k + s];
    __syncthreads();
  }
  const float n  = red[0];
  const float cs = (ncs + EPS_F) / (n + (float)K * EPS_F) * n;
  #pragma unroll
  for (int d = 0; d < D_DIM; ++d) {
    const float ea = embed_avg[(size_t)d * K + k] * DECAY_F + OMD_F * embed_sum[(size_t)d * K + k];
    out_ea[(size_t)d * K + k]     = ea;
    out_ew[(size_t)k * D_DIM + d] = ea / cs;
  }
  if (k == 0) out_diff[0] = 1.25f * sqsum[0] * inv_count;
}

// --------------------------------------------------------------- launch -----
extern "C" void kernel_launch(void* const* d_in, const int* in_sizes, int n_in,
                              void* d_out, int out_size, void* d_ws, size_t ws_size,
                              hipStream_t stream) {
  const float* z_e          = (const float*)d_in[0];
  const float* embed_w      = (const float*)d_in[1];
  const float* cluster_size = (const float*)d_in[2];
  const float* embed_avg    = (const float*)d_in[3];

  const int D = D_DIM;
  const int K = in_sizes[2];        // 1024
  const int N = in_sizes[0] / D;    // 32768

  float* out = (float*)d_out;
  const size_t zq_elems = (size_t)N * D;
  float* out_zq   = out;
  float* out_diff = out + zq_elems;
  float* out_ind  = out_diff + 1;
  float* out_cs   = out_ind + N;
  float* out_ea   = out_cs + K;
  float* out_ew   = out_ea + (size_t)D * K;

  __bf16* ewHi     = (__bf16*)d_ws;
  __bf16* ewLo     = ewHi + (size_t)K * D;
  float*  codeNorm = (float*)(ewLo + (size_t)K * D);
  float*  onehot   = codeNorm + K;
  float*  embedSum = onehot + K;
  float*  sqsum    = embedSum + (size_t)D * K;

  const int zero_count = K + D * K + 1;  // onehot + embed_sum + sqsum (contiguous)
  vq_init<<<(zero_count + 255) / 256, 256, 0, stream>>>(onehot, zero_count);
  vq_prep<<<(K + 7) / 8, 256, 0, stream>>>(embed_w, ewHi, ewLo, codeNorm, K);
  vq_main<<<N / 128, 256, 0, stream>>>(z_e, embed_w, ewHi, ewLo, codeNorm,
                                       out_zq, out_ind, onehot, embedSum, sqsum, K);
  vq_finalize<<<1, K, 0, stream>>>(cluster_size, embed_avg, onehot, embedSum, sqsum,
                                   out_cs, out_ea, out_ew, out_diff, K,
                                   1.0f / (float)zq_elems);
}
